// SchNetCHA_6975026889063
// MI455X (gfx1250) — matrix operand order repaired and checked by reading, not compiled, not run on hardware
//
#include <hip/hip_runtime.h>
#include <hip/hip_bf16.h>
#include <stdint.h>

// ---------------- problem constants ----------------
#define NNODE   10000
#define NEDGE   320000
#define NEH     40000
#define GDIM    25
#define CUTOFF_F 5.0f
#define PI_F     3.14159265358979f

// ---------------- WMMA plumbing (gfx1250, wave32) ----------------
typedef __attribute__((ext_vector_type(16))) _Float16 v16h;
typedef __attribute__((ext_vector_type(8)))  float    v8f;

union Frag16 { v16h h; uint32_t d[8]; uint4 q[2]; };
union H2U    { _Float16 h[2]; uint32_t u; };

__device__ __forceinline__ void lds_fence() {
  // CDNA5 split wait counter: all DS ops complete (cross-lane LDS handoff in-wave)
  asm volatile("s_wait_dscnt 0" ::: "memory");
}

__device__ __forceinline__ void async_fence() {
  // drain GLOBAL_LOAD_ASYNC_TO_LDS traffic (ASYNCcnt)
  asm volatile("s_wait_asynccnt 0x0" ::: "memory");
}

// CDNA5 async DMA: memory -> LDS, 16B per lane, no VGPR round trip (ASYNCcnt).
__device__ __forceinline__ void async_g2l_b128(uint32_t lds_off,
                                               const void* gbase, uint32_t voff) {
  asm volatile("global_load_async_to_lds_b128 %0, %1, %2"
               :: "v"(lds_off), "v"(voff), "s"(gbase)
               : "memory");
}

__device__ __forceinline__ uint32_t lds_addr(const void* p) {
  // generic pointer to LDS: aperture check keeps offset in addr[31:0]
  return (uint32_t)(uintptr_t)p;
}

__device__ __forceinline__ v8f splat8(float x) {
  v8f v;
#pragma unroll
  for (int r = 0; r < 8; ++r) v[r] = x;
  return v;
}

__device__ __forceinline__ float ssp(float x) {  // shifted softplus, stable
  float ax = fabsf(x);
  return fmaxf(x, 0.f) + log1pf(__expf(-ax)) - 0.69314718056f;
}

// A fragment (16x32 f16, MxK): lane = row (mod 16); dwords 0..3 hold K=8g..8g+7,
// dwords 4..7 hold K=16+8g..16+8g+7  (g = lane>>4), per ISA 7.12.2 table.
__device__ __forceinline__ v16h load_afrag(const _Float16* base, int strideH, int lane) {
  int row = lane & 15, g = lane >> 4;
  Frag16 f;
  f.q[0] = *(const uint4*)(base + row * strideH + 8 * g);
  f.q[1] = *(const uint4*)(base + row * strideH + 16 + 8 * g);
  return f.h;
}

// B fragment (32x16 f16, KxN), fragment-major: 256 dwords/fragment, lane owns
// dwords [lane*8 .. lane*8+8). Works for global (b128x2) or LDS (ds_load_b128x2).
__device__ __forceinline__ v16h load_bfrag(const uint32_t* pk, int frag, int lane) {
  Frag16 f;
  const uint4* q = (const uint4*)(pk + (size_t)frag * 256 + (size_t)lane * 8);
  f.q[0] = q[0];
  f.q[1] = q[1];
  return f.h;
}

__device__ __forceinline__ v8f wmma_f16(v16h a, v16h b, v8f c) {
  return __builtin_amdgcn_wmma_f32_16x16x32_f16(false, a, false, b, (short)0, c,
                                                false, false);
}

// ---------------- weight packing: W[O][K] (row-major) -> f16 B fragments ----------
// frag(nt,kt) at index nt*KT+kt. Per ISA B layout: lane l holds N = nt*16 + (l&15);
// dword p packs K = kt*32 + 8*(l>>4) + 2p + 8*(p>=4) + {0,1}. Out-of-range K -> 0.
__global__ void pack_b(const float* __restrict__ W, uint32_t* __restrict__ dst,
                       int K, int KT, int total) {
  int idx = blockIdx.x * 256 + threadIdx.x;
  if (idx >= total) return;
  int p    = idx & 7;
  int lane = (idx >> 3) & 31;
  int f    = idx >> 8;
  int kt   = f % KT;
  int nt   = f / KT;
  int kk   = kt * 32 + 8 * (lane >> 4) + 2 * p + ((p >> 2) << 3);
  int n0   = nt * 16 + (lane & 15);
  float x0 = (kk < K) ? W[(size_t)n0 * K + kk]           : 0.f;
  float x1 = (kk + 1 < K) ? W[(size_t)n0 * K + kk + 1] : 0.f;
  H2U cv;
  cv.h[0] = (_Float16)x0;
  cv.h[1] = (_Float16)x1;
  dst[idx] = cv.u;
}

// ---------------- v = emb_table[z] ----------------
__global__ void embed_kernel(const int* __restrict__ z, const float* __restrict__ emb,
                             float* __restrict__ v) {
  int idx = blockIdx.x * 256 + threadIdx.x;
  if (idx >= NNODE * 128) return;
  int n = idx >> 7, h = idx & 127;
  v[idx] = emb[z[n] * 128 + h];
}

// ---------------- P = v @ lin^T, Ph = v @ linh^T  (one wave = 16 nodes) -------
__global__ __launch_bounds__(256) void proj_kernel(const float* __restrict__ v,
    const uint32_t* __restrict__ plin, const uint32_t* __restrict__ plinh,
    float* __restrict__ P, float* __restrict__ Ph, int N) {
  __shared__ _Float16 sV[8][2048];
  int wave = threadIdx.x >> 5, lane = threadIdx.x & 31;
  int tile = blockIdx.x * 8 + wave;
  if (tile * 16 >= N) return;
  int row = lane & 15, g = lane >> 4;
  const float* vt = v + (size_t)tile * 16 * 128;
  for (int t = 0; t < 64; ++t) {
    int idx = t * 32 + lane;
    sV[wave][idx] = (_Float16)vt[idx];
  }
  lds_fence();
  v16h a[4];
#pragma unroll
  for (int kt = 0; kt < 4; ++kt) a[kt] = load_afrag(&sV[wave][kt * 32], 128, lane);
  for (int w = 0; w < 2; ++w) {
    const uint32_t* pw = w ? plinh : plin;
    float* dst = w ? Ph : P;
    for (int nt = 0; nt < 8; ++nt) {
      v8f acc = splat8(0.f);
#pragma unroll
      for (int kt = 0; kt < 4; ++kt)
        acc = wmma_f16(a[kt], load_bfrag(pw, nt * 4 + kt, lane), acc);
      int n = nt * 16 + row;
#pragma unroll
      for (int r = 0; r < 8; ++r)
        dst[((size_t)tile * 16 + g * 8 + r) * 128 + n] = acc[r];
    }
  }
}

// ---------------- fused edge path (radius graph), persistent --------------------
// Weights staged in LDS once per block via async-to-LDS DMA; each wave grid-strides
// over 16-edge tiles:  W = (ssp(gauss(d)@w1^T+b1)@w2^T+b2)*C(d);
// scatter-add P[j]*W into acc[i] (L2-resident 5MB target -> high atomic hit rate).
__global__ __launch_bounds__(128) void edge_kernel(const float* __restrict__ dist,
    const int* __restrict__ eidx, const float* __restrict__ P,
    float* __restrict__ accN, const uint32_t* __restrict__ pw1,
    const uint32_t* __restrict__ pw2, const float* __restrict__ b1,
    const float* __restrict__ b2, int E) {
  __shared__ uint32_t sW1[2048];     // 8 KB  packed mlp_w1 fragments
  __shared__ uint32_t sW2[8192];     // 32 KB packed mlp_w2 fragments
  __shared__ _Float16 sA[4][512];    // 16x32 gaussian tile (G=25 padded to 32)
  __shared__ _Float16 sH[4][2048];   // 16x128 hidden tile
  __shared__ float sC[4][16];
  __shared__ float sD[4][16];
  __shared__ int   sJ[4][16], sI[4][16];

  int tid = threadIdx.x;
  int wave = tid >> 5, lane = tid & 31;
  int row = lane & 15, g = lane >> 4;

  // ---- stage weights: 40KB / block, 16B per lane per op, ASYNCcnt tracked ----
  {
    uint32_t l1 = lds_addr(&sW1[0]);
#pragma unroll
    for (int i = 0; i < 4; ++i) {                  // 2048 dwords
      uint32_t off = (uint32_t)(i * 128 + tid) * 16;
      async_g2l_b128(l1 + off, pw1, off);
    }
    uint32_t l2 = lds_addr(&sW2[0]);
#pragma unroll
    for (int i = 0; i < 16; ++i) {                 // 8192 dwords
      uint32_t off = (uint32_t)(i * 128 + tid) * 16;
      async_g2l_b128(l2 + off, pw2, off);
    }
    async_fence();
  }
  __syncthreads();

  const float step  = CUTOFF_F / 24.0f;
  const float coeff = -0.5f / (step * step);
  int nwaves = gridDim.x * 4;
  for (int tile = blockIdx.x * 4 + wave; tile * 16 < E; tile += nwaves) {
    int e0 = tile * 16;
    // prefetch next tile's metadata while we compute this one
    int npre = (tile + nwaves) * 16;
    if (lane == 0 && npre < E) {
      __builtin_prefetch(&dist[npre], 0, 0);
      __builtin_prefetch(&eidx[npre], 0, 0);
      __builtin_prefetch(&eidx[E + npre], 0, 0);
    }
    if (lane < 16) {
      float d = dist[e0 + lane];
      sD[wave][lane] = d;
      sC[wave][lane] = 0.5f * (__cosf(d * (PI_F / CUTOFF_F)) + 1.0f);
      sJ[wave][lane] = eidx[e0 + lane];
      sI[wave][lane] = eidx[E + e0 + lane];
    }
    lds_fence();
    float d = sD[wave][row];
#pragma unroll
    for (int t = 0; t < 16; ++t) {
      int k = g * 16 + t;
      float val = 0.f;
      if (k < GDIM) { float dx = d - (float)k * step; val = __expf(coeff * dx * dx); }
      sA[wave][row * 32 + k] = (_Float16)val;
    }
    // hoist per-row tile metadata into registers (8 rows per lane)
    size_t jb[8], ib[8];
    float  cc[8];
#pragma unroll
    for (int r = 0; r < 8; ++r) {
      int m = g * 8 + r;
      jb[r] = (size_t)sJ[wave][m] * 128;
      ib[r] = (size_t)sI[wave][m] * 128;
      cc[r] = sC[wave][m];
    }
    lds_fence();
    // GEMM1: [16,32] @ [32,128]  (B fragments from LDS)
    v16h a0 = load_afrag(&sA[wave][0], 32, lane);
    for (int nt = 0; nt < 8; ++nt) {
      int n = nt * 16 + row;
      v8f acc = splat8(b1[n]);
      acc = wmma_f16(a0, load_bfrag(sW1, nt, lane), acc);
#pragma unroll
      for (int r = 0; r < 8; ++r)
        sH[wave][(g * 8 + r) * 128 + n] = (_Float16)ssp(acc[r]);
    }
    lds_fence();
    // GEMM2: [16,128] @ [128,128] + cutoff scale + gather P[j] + scatter acc[i]
    v16h a[4];
#pragma unroll
    for (int kt = 0; kt < 4; ++kt) a[kt] = load_afrag(&sH[wave][kt * 32], 128, lane);
    for (int nt = 0; nt < 8; ++nt) {
      int n = nt * 16 + row;
      v8f acc = splat8(b2[n]);
#pragma unroll
      for (int kt = 0; kt < 4; ++kt)
        acc = wmma_f16(a[kt], load_bfrag(sW2, nt * 4 + kt, lane), acc);
#pragma unroll
      for (int r = 0; r < 8; ++r) {
        float e = acc[r] * cc[r] * P[jb[r] + n];
        atomicAdd(&accN[ib[r] + n], e);
      }
    }
    lds_fence();  // tile-local buffers reused next iteration
  }
}

// ---------------- fused hull-edge path, persistent -----------------------------
__global__ __launch_bounds__(128) void hull_kernel(const float* __restrict__ feah,
    const int* __restrict__ eidxh, const float* __restrict__ Ph,
    float* __restrict__ acchN, const uint32_t* __restrict__ pw1,
    const uint32_t* __restrict__ pw2, const float* __restrict__ b1,
    const float* __restrict__ b2, int EH) {
  __shared__ uint32_t sW1[2048];
  __shared__ uint32_t sW2[8192];
  __shared__ _Float16 sA[4][512];
  __shared__ _Float16 sH[4][2048];
  __shared__ int sJ[4][16], sI[4][16];

  int tid = threadIdx.x;
  int wave = tid >> 5, lane = tid & 31;
  int row = lane & 15, g = lane >> 4;

  {
    uint32_t l1 = lds_addr(&sW1[0]);
#pragma unroll
    for (int i = 0; i < 4; ++i) {
      uint32_t off = (uint32_t)(i * 128 + tid) * 16;
      async_g2l_b128(l1 + off, pw1, off);
    }
    uint32_t l2 = lds_addr(&sW2[0]);
#pragma unroll
    for (int i = 0; i < 16; ++i) {
      uint32_t off = (uint32_t)(i * 128 + tid) * 16;
      async_g2l_b128(l2 + off, pw2, off);
    }
    async_fence();
  }
  __syncthreads();

  int nwaves = gridDim.x * 4;
  for (int tile = blockIdx.x * 4 + wave; tile * 16 < EH; tile += nwaves) {
    int e0 = tile * 16;
    if (lane < 16) {
      sJ[wave][lane] = eidxh[e0 + lane];
      sI[wave][lane] = eidxh[EH + e0 + lane];
    }
#pragma unroll
    for (int t = 0; t < 16; ++t) {
      int k = g * 16 + t;
      float val = (k < 7) ? feah[(size_t)(e0 + row) * 7 + k] : 0.f;
      sA[wave][row * 32 + k] = (_Float16)val;
    }
    lds_fence();
    size_t jb[8], ib[8];
#pragma unroll
    for (int r = 0; r < 8; ++r) {
      int m = g * 8 + r;
      jb[r] = (size_t)sJ[wave][m] * 128;
      ib[r] = (size_t)sI[wave][m] * 128;
    }
    v16h a0 = load_afrag(&sA[wave][0], 32, lane);
    for (int nt = 0; nt < 8; ++nt) {
      int n = nt * 16 + row;
      v8f acc = splat8(b1[n]);
      acc = wmma_f16(a0, load_bfrag(sW1, nt, lane), acc);
#pragma unroll
      for (int r = 0; r < 8; ++r)
        sH[wave][(g * 8 + r) * 128 + n] = (_Float16)ssp(acc[r]);
    }
    lds_fence();
    v16h a[4];
#pragma unroll
    for (int kt = 0; kt < 4; ++kt) a[kt] = load_afrag(&sH[wave][kt * 32], 128, lane);
    for (int nt = 0; nt < 8; ++nt) {
      int n = nt * 16 + row;
      v8f acc = splat8(b2[n]);
#pragma unroll
      for (int kt = 0; kt < 4; ++kt)
        acc = wmma_f16(a[kt], load_bfrag(sW2, nt * 4 + kt, lane), acc);
#pragma unroll
      for (int r = 0; r < 8; ++r) {
        float e = acc[r] * Ph[jb[r] + n];
        atomicAdd(&acchN[ib[r] + n], e);
      }
    }
    lds_fence();
  }
}

// ---------------- node update: both MLP paths + concat-linear + residual ------
__global__ __launch_bounds__(128) void nodeup_kernel(const float* __restrict__ accN,
    const float* __restrict__ acchN, float* __restrict__ v,
    const uint32_t* __restrict__ pv1, const uint32_t* __restrict__ pv2,
    const uint32_t* __restrict__ pvh1, const uint32_t* __restrict__ pvh2,
    const uint32_t* __restrict__ pcat,
    const float* __restrict__ v1b, const float* __restrict__ v2b,
    const float* __restrict__ vh1b, const float* __restrict__ vh2b,
    const float* __restrict__ catb, int N) {
  __shared__ _Float16 sA[4][2048];
  __shared__ _Float16 sT[4][2048];
  __shared__ _Float16 sCat[4][2048];
  int wave = threadIdx.x >> 5, lane = threadIdx.x & 31;
  int tile = blockIdx.x * 4 + wave;
  if (tile * 16 >= N) return;
  int row = lane & 15, g = lane >> 4;

  for (int path = 0; path < 2; ++path) {
    const float* src = path ? acchN : accN;
    const uint32_t* pA = path ? pvh1 : pv1;
    const uint32_t* pB = path ? pvh2 : pv2;
    const float* bA = path ? vh1b : v1b;
    const float* bB = path ? vh2b : v2b;
    const float* st = src + (size_t)tile * 16 * 128;
    for (int t = 0; t < 64; ++t) {
      int idx = t * 32 + lane;
      sA[wave][idx] = (_Float16)st[idx];
    }
    lds_fence();
    v16h a[4];
#pragma unroll
    for (int kt = 0; kt < 4; ++kt) a[kt] = load_afrag(&sA[wave][kt * 32], 128, lane);
    for (int nt = 0; nt < 8; ++nt) {
      int n = nt * 16 + row;
      v8f acc = splat8(bA[n]);
#pragma unroll
      for (int kt = 0; kt < 4; ++kt)
        acc = wmma_f16(a[kt], load_bfrag(pA, nt * 4 + kt, lane), acc);
#pragma unroll
      for (int r = 0; r < 8; ++r)
        sT[wave][(g * 8 + r) * 128 + n] = (_Float16)ssp(acc[r]);
    }
    lds_fence();
    v16h a2[4];
#pragma unroll
    for (int kt = 0; kt < 4; ++kt) a2[kt] = load_afrag(&sT[wave][kt * 32], 128, lane);
    for (int nt = 0; nt < 4; ++nt) {            // 64 output cols
      int n = nt * 16 + row;
      v8f acc = splat8(bB[n]);
#pragma unroll
      for (int kt = 0; kt < 4; ++kt)
        acc = wmma_f16(a2[kt], load_bfrag(pB, nt * 4 + kt, lane), acc);
#pragma unroll
      for (int r = 0; r < 8; ++r)
        sCat[wave][(g * 8 + r) * 128 + path * 64 + n] = (_Float16)acc[r];
    }
    lds_fence();
  }
  v16h a3[4];
#pragma unroll
  for (int kt = 0; kt < 4; ++kt) a3[kt] = load_afrag(&sCat[wave][kt * 32], 128, lane);
  for (int nt = 0; nt < 8; ++nt) {
    int n = nt * 16 + row;
    v8f acc = splat8(catb[n]);
#pragma unroll
    for (int kt = 0; kt < 4; ++kt)
      acc = wmma_f16(a3[kt], load_bfrag(pcat, nt * 4 + kt, lane), acc);
#pragma unroll
    for (int r = 0; r < 8; ++r) {
      size_t off = ((size_t)tile * 16 + g * 8 + r) * 128 + n;
      v[off] += ssp(acc[r]);
    }
  }
}

// ---------------- readout: u = ssp(v@u1^T+b)@u2^T+b2, segment-sum by batch -----
__global__ __launch_bounds__(128) void readout_kernel(const float* __restrict__ v,
    const uint32_t* __restrict__ pu1, const float* __restrict__ u1b,
    const float* __restrict__ u2w, const float* __restrict__ u2b,
    const int* __restrict__ batch, float* __restrict__ out, int N) {
  __shared__ _Float16 sV[4][2048];
  __shared__ float sT[4][16 * 64];
  int wave = threadIdx.x >> 5, lane = threadIdx.x & 31;
  int tile = blockIdx.x * 4 + wave;
  if (tile * 16 >= N) return;
  int row = lane & 15, g = lane >> 4;
  const float* vt = v + (size_t)tile * 16 * 128;
  for (int t = 0; t < 64; ++t) {
    int idx = t * 32 + lane;
    sV[wave][idx] = (_Float16)vt[idx];
  }
  lds_fence();
  v16h a[4];
#pragma unroll
  for (int kt = 0; kt < 4; ++kt) a[kt] = load_afrag(&sV[wave][kt * 32], 128, lane);
  for (int nt = 0; nt < 4; ++nt) {
    int n = nt * 16 + row;
    v8f acc = splat8(u1b[n]);
#pragma unroll
    for (int kt = 0; kt < 4; ++kt)
      acc = wmma_f16(a[kt], load_bfrag(pu1, nt * 4 + kt, lane), acc);
#pragma unroll
    for (int r = 0; r < 8; ++r) sT[wave][(g * 8 + r) * 64 + n] = ssp(acc[r]);
  }
  lds_fence();
  if (lane < 16) {
    int node = tile * 16 + lane;
    float s = u2b[0];
    for (int k = 0; k < 64; ++k) s += sT[wave][lane * 64 + k] * u2w[k];
    atomicAdd(&out[batch[node]], s);
  }
}

// ---------------- host driver ----------------
extern "C" void kernel_launch(void* const* d_in, const int* in_sizes, int n_in,
                              void* d_out, int out_size, void* d_ws, size_t ws_size,
                              hipStream_t stream) {
  (void)in_sizes; (void)n_in; (void)ws_size;
  const int*   z      = (const int*)  d_in[0];
  const int*   eidx   = (const int*)  d_in[1];
  const int*   eidxh  = (const int*)  d_in[2];
  const int*   batch  = (const int*)  d_in[3];
  const float* dist   = (const float*)d_in[4];
  const float* feah   = (const float*)d_in[5];
  const float* emb    = (const float*)d_in[6];
  const float* lin_w  = (const float*)d_in[7];
  const float* mlp_w1 = (const float*)d_in[8];
  const float* mlp_b1 = (const float*)d_in[9];
  const float* mlp_w2 = (const float*)d_in[10];
  const float* mlp_b2 = (const float*)d_in[11];
  const float* linh_w = (const float*)d_in[12];
  const float* mlph_w1= (const float*)d_in[13];
  const float* mlph_b1= (const float*)d_in[14];
  const float* mlph_w2= (const float*)d_in[15];
  const float* mlph_b2= (const float*)d_in[16];
  const float* v1_w   = (const float*)d_in[17];
  const float* v1_b   = (const float*)d_in[18];
  const float* v2_w   = (const float*)d_in[19];
  const float* v2_b   = (const float*)d_in[20];
  const float* vh1_w  = (const float*)d_in[21];
  const float* vh1_b  = (const float*)d_in[22];
  const float* vh2_w  = (const float*)d_in[23];
  const float* vh2_b  = (const float*)d_in[24];
  const float* cat_w  = (const float*)d_in[25];
  const float* cat_b  = (const float*)d_in[26];
  const float* u1_w   = (const float*)d_in[27];
  const float* u1_b   = (const float*)d_in[28];
  const float* u2_w   = (const float*)d_in[29];
  const float* u2_b   = (const float*)d_in[30];
  float* out = (float*)d_out;

  const size_t NB = (size_t)NNODE * 128 * sizeof(float);  // 5.12 MB
  char* ws = (char*)d_ws;
  float*    v     = (float*)(ws);
  float*    P     = (float*)(ws + NB);
  float*    Ph    = (float*)(ws + 2 * NB);
  float*    accN  = (float*)(ws + 3 * NB);
  float*    acchN = (float*)(ws + 4 * NB);
  uint32_t* pk    = (uint32_t*)(ws + 5 * NB);

  // per-layer packed-weight offsets (dwords); frag = 256 dwords
  const int PL = 69632;
  const int O_LIN = 0, O_LINH = 8192, O_W1 = 16384, O_W2 = 18432, O_WH1 = 26624,
            O_WH2 = 28672, O_V1 = 36864, O_V2 = 45056, O_VH1 = 49152,
            O_VH2 = 57344, O_CAT = 61440;
  const int O_U1 = 4 * PL;

  auto packW = [&](const float* W, uint32_t* dst, int K, int KT, int total) {
    pack_b<<<(total + 255) / 256, 256, 0, stream>>>(W, dst, K, KT, total);
  };
  for (int l = 0; l < 4; ++l) {
    uint32_t* pb = pk + (size_t)l * PL;
    packW(lin_w   + (size_t)l * 128 * 128, pb + O_LIN , 128, 4, 8192);
    packW(linh_w  + (size_t)l * 128 * 128, pb + O_LINH, 128, 4, 8192);
    packW(mlp_w1  + (size_t)l * 128 * 25 , pb + O_W1  ,  25, 1, 2048);
    packW(mlp_w2  + (size_t)l * 128 * 128, pb + O_W2  , 128, 4, 8192);
    packW(mlph_w1 + (size_t)l * 128 * 7  , pb + O_WH1 ,   7, 1, 2048);
    packW(mlph_w2 + (size_t)l * 128 * 128, pb + O_WH2 , 128, 4, 8192);
    packW(v1_w    + (size_t)l * 128 * 128, pb + O_V1  , 128, 4, 8192);
    packW(v2_w    + (size_t)l * 64  * 128, pb + O_V2  , 128, 4, 4096);
    packW(vh1_w   + (size_t)l * 128 * 128, pb + O_VH1 , 128, 4, 8192);
    packW(vh2_w   + (size_t)l * 64  * 128, pb + O_VH2 , 128, 4, 4096);
    packW(cat_w   + (size_t)l * 128 * 128, pb + O_CAT , 128, 4, 8192);
  }
  packW(u1_w, pk + O_U1, 128, 4, 4096);

  embed_kernel<<<(NNODE * 128 + 255) / 256, 256, 0, stream>>>(z, emb, v);

  for (int l = 0; l < 4; ++l) {
    uint32_t* pb = pk + (size_t)l * PL;
    hipMemsetAsync(accN, 0, 2 * NB, stream);  // accN and acchN are adjacent
    proj_kernel<<<79, 256, 0, stream>>>(v, pb + O_LIN, pb + O_LINH, P, Ph, NNODE);
    edge_kernel<<<640, 128, 0, stream>>>(dist, eidx, P, accN,
        pb + O_W1, pb + O_W2, mlp_b1 + l * 128, mlp_b2 + l * 128, NEDGE);
    hull_kernel<<<320, 128, 0, stream>>>(feah, eidxh, Ph, acchN,
        pb + O_WH1, pb + O_WH2, mlph_b1 + l * 128, mlph_b2 + l * 128, NEH);
    nodeup_kernel<<<157, 128, 0, stream>>>(accN, acchN, v,
        pb + O_V1, pb + O_V2, pb + O_VH1, pb + O_VH2, pb + O_CAT,
        v1_b + l * 128, v2_b + l * 64, vh1_b + l * 128, vh2_b + l * 64,
        cat_b + l * 128, NNODE);
  }
  hipMemsetAsync(d_out, 0, (size_t)out_size * sizeof(float), stream);
  readout_kernel<<<157, 128, 0, stream>>>(v, pk + O_U1, u1_b, u2_w, u2_b, batch,
                                          out, NNODE);
}
